// CQAttention_25744033973049
// MI455X (gfx1250) — matrix operand order repaired and checked by reading, not compiled, not run on hardware
//
#include <hip/hip_runtime.h>
#include <hip/hip_bf16.h>

// CQAttention for MI455X (gfx1250):
//  - all GEMMs on v_wmma_f32_16x16x32_bf16
//  - B operands stored pre-swizzled in WMMA fragment layout (2x b128 per frag)
//  - dominant GEMM (Bm = T @ Ct) stages its B panel in LDS via the Tensor Data
//    Mover (tensor_load_to_lds + s_wait_tensorcnt), fragments read via ds_load_b128
//  - softmax / small reductions in f32 VALU with wave32 shuffles

#define B_  32
#define H_  768
#define LC  512
#define LQ  128
#define NEG_INF_F (-1e30f)

typedef __bf16 bf16;
typedef __attribute__((ext_vector_type(16))) __bf16 v16bf;
typedef __attribute__((ext_vector_type(8)))  float  v8f;
typedef unsigned int u32x4 __attribute__((ext_vector_type(4)));
typedef int          i32x4 __attribute__((ext_vector_type(4)));
typedef int          i32x8 __attribute__((ext_vector_type(8)));

// ---------------- wave32 reductions ----------------
__device__ __forceinline__ float wave_sum(float v) {
    #pragma unroll
    for (int o = 16; o; o >>= 1) v += __shfl_xor(v, o, 32);
    return v;
}
__device__ __forceinline__ float wave_max(float v) {
    #pragma unroll
    for (int o = 16; o; o >>= 1) v = fmaxf(v, __shfl_xor(v, o, 32));
    return v;
}

// ---------------- WMMA fragment helpers ----------------
// A (16x32 bf16, row-major MxK): lanes 0-15 -> M=lane, K {0..15};
// lanes 16-31 -> M=lane-16, K {16..31}. Two 16B contiguous loads.
__device__ __forceinline__ v16bf load_a_frag(const bf16* __restrict__ A, int lda,
                                             int m0, int k0) {
    const int lane = threadIdx.x & 31;
    const int m  = m0 + (lane & 15);
    const int kb = (lane >> 4) * 16;
    const bf16* p = A + (size_t)m * lda + k0 + kb;
    union { u32x4 u[2]; v16bf v; } f;
    f.u[0] = *(const u32x4*)(p);
    f.u[1] = *(const u32x4*)(p + 8);
    return f.v;
}

// Swizzled B layout: (K x N) matrix stored as 32x16 tiles of 512 elements,
// kt-outer:  tile_base = ((k>>5)*(N>>4) + (n>>4))*512
// in-tile :  lane = ((k>>4)&1)*16 + (n&15), idx = k&15  ->  lane*16 + idx
// A fragment for lane l is 16 contiguous bf16 at tile_base + l*16 (32 bytes).
__device__ __forceinline__ size_t bswz(int k, int n, int N) {
    return ((size_t)(k >> 5) * (N >> 4) + (n >> 4)) * 512
         + (size_t)((((k >> 4) & 1) * 16 + (n & 15)) * 16 + (k & 15));
}

__device__ __forceinline__ v16bf load_b_frag_sw(const bf16* __restrict__ B, int N,
                                                int k0, int n0) {
    const int lane = threadIdx.x & 31;
    const bf16* p = B + ((size_t)(k0 >> 5) * (N >> 4) + (n0 >> 4)) * 512 + lane * 16;
    union { u32x4 u[2]; v16bf v; } f;
    f.u[0] = *(const u32x4*)(p);
    f.u[1] = *(const u32x4*)(p + 8);
    return f.v;
}

// B fragment from an LDS-staged panel: panel element (kt, nt_local, l*16+i)
// at byte offset ((kt*4 + nt_local)*512 + l*16)*2.
__device__ __forceinline__ v16bf load_b_frag_lds(const bf16* sm, int kt, int t) {
    const int lane = threadIdx.x & 31;
    const bf16* p = sm + ((size_t)(kt * 4 + t) * 512) + lane * 16;
    union { u32x4 u[2]; v16bf v; } f;
    f.u[0] = *(const u32x4*)(p);
    f.u[1] = *(const u32x4*)(p + 8);
    return f.v;
}

#define WMMA_BF16(a, b, c) \
    __builtin_amdgcn_wmma_f32_16x16x32_bf16(false, (a), false, (b), (short)0, (c), false, false)

// ---------------- conversion / small kernels ----------------
__global__ void convC_kernel(const float* __restrict__ C, const float* __restrict__ w4mlu,
                             bf16* __restrict__ Ctb, bf16* __restrict__ Csb) {
    size_t idx = (size_t)blockIdx.x * blockDim.x + threadIdx.x;
    if (idx >= (size_t)B_ * H_ * LC) return;
    int c = (int)(idx % LC);
    int h = (int)((idx / LC) % H_);
    int b = (int)(idx / ((size_t)H_ * LC));
    float v = C[idx];
    // Ct as swizzled B operand of Bm GEMM: (K=Lc, N=H), element (k=c, n=h)
    Ctb[(size_t)b * LC * H_ + bswz(c, h, H_)] = (bf16)v;
    // Ct*w4mlu as row-major A operand of S GEMM
    Csb[((size_t)b * LC + c) * H_ + h] = (bf16)(v * w4mlu[h]);
}

__global__ void convQ_kernel(const float* __restrict__ Q,
                             bf16* __restrict__ Qb, bf16* __restrict__ Qtb) {
    size_t idx = (size_t)blockIdx.x * blockDim.x + threadIdx.x;
    if (idx >= (size_t)B_ * H_ * LQ) return;
    int q = (int)(idx % LQ);
    int h = (int)((idx / LQ) % H_);
    int b = (int)(idx / ((size_t)H_ * LQ));
    float v = Q[idx];
    // Q as swizzled B of S GEMM: (K=H, N=Lq), element (k=h, n=q)
    Qb[(size_t)b * H_ * LQ + bswz(h, q, LQ)] = (bf16)v;
    // Qt as swizzled B of A GEMM: (K=Lq, N=H), element (k=q, n=h)
    Qtb[(size_t)b * LQ * H_ + bswz(q, h, H_)] = (bf16)v;
}

__global__ void ks0_kernel(const float* __restrict__ C, const float* __restrict__ w4C,
                           float* __restrict__ s0) {
    int row = blockIdx.x * 8 + (threadIdx.x >> 5);       // b*LC + c
    int lane = threadIdx.x & 31;
    int b = row / LC, c = row % LC;
    float acc = 0.f;
    for (int h = lane; h < H_; h += 32)
        acc += C[((size_t)b * H_ + h) * LC + c] * w4C[h];
    acc = wave_sum(acc);
    if (lane == 0) s0[row] = acc;
}

__global__ void ks1_kernel(const float* __restrict__ Q, const float* __restrict__ w4Q,
                           float* __restrict__ s1) {
    int row = blockIdx.x * 8 + (threadIdx.x >> 5);       // b*LQ + q
    int lane = threadIdx.x & 31;
    int b = row / LQ, q = row % LQ;
    float acc = 0.f;
    for (int h = lane; h < H_; h += 32)
        acc += Q[((size_t)b * H_ + h) * LQ + q] * w4Q[h];
    acc = wave_sum(acc);
    if (lane == 0) s1[row] = acc;
}

// ---------------- S = Cs @ Q + s0 + s1 + bias ----------------
__global__ void gemmS_kernel(const bf16* __restrict__ Cs, const bf16* __restrict__ Qb,
                             const float* __restrict__ s0, const float* __restrict__ s1,
                             const float* __restrict__ bias, float* __restrict__ S) {
    const int b = blockIdx.z;
    const int m0 = blockIdx.x * 64;
    const int n0 = blockIdx.y * 64;
    const int wave = threadIdx.x >> 5;
    const int lane = threadIdx.x & 31;
    const bf16* A  = Cs + (size_t)b * LC * H_;
    const bf16* Bm = Qb + (size_t)b * H_ * LQ;
    const int mw = m0 + wave * 16;

    v8f acc[4] = {};
    for (int k0 = 0; k0 < H_; k0 += 32) {
        if (k0 + 32 < H_)
            __builtin_prefetch(A + (size_t)mw * H_ + k0 + 32, 0, 0);
        v16bf a = load_a_frag(A, H_, mw, k0);
        #pragma unroll
        for (int t = 0; t < 4; ++t) {
            v16bf bb = load_b_frag_sw(Bm, LQ, k0, n0 + t * 16);
            acc[t] = WMMA_BF16(a, bb, acc[t]);
        }
    }
    const float bv = bias[0];
    const int mh = (lane >> 4) * 8, nl = lane & 15;
    #pragma unroll
    for (int t = 0; t < 4; ++t) {
        #pragma unroll
        for (int r = 0; r < 8; ++r) {
            int m = mw + r + mh;
            int n = n0 + t * 16 + nl;
            S[((size_t)b * LC + m) * LQ + n] = acc[t][r] + s0[b * LC + m] + s1[b * LQ + n] + bv;
        }
    }
}

// ---------------- softmax over q (axis 2), masked by Qmask ----------------
__global__ void rowsm_kernel(const float* __restrict__ S, const float* __restrict__ Qmask,
                             bf16* __restrict__ S1) {
    int row = blockIdx.x * 8 + (threadIdx.x >> 5);       // b*LC + c
    int lane = threadIdx.x & 31;
    int b = row / LC;
    const float* s = S + (size_t)row * LQ;
    float v[4], mx = NEG_INF_F;
    #pragma unroll
    for (int i = 0; i < 4; ++i) {
        int q = lane + 32 * i;
        float qm = Qmask[b * LQ + q];
        v[i] = s[q] * qm + (1.f - qm) * NEG_INF_F;
        mx = fmaxf(mx, v[i]);
    }
    mx = wave_max(mx);
    float sum = 0.f;
    #pragma unroll
    for (int i = 0; i < 4; ++i) { v[i] = __expf(v[i] - mx); sum += v[i]; }
    sum = wave_sum(sum);
    float inv = 1.f / sum;
    bf16* o = S1 + (size_t)row * LQ;                      // row-major A operand
    #pragma unroll
    for (int i = 0; i < 4; ++i) o[lane + 32 * i] = (bf16)(v[i] * inv);
}

// -------- softmax over c (axis 1), masked by Cmask; stored as swizzled B -------
__global__ void colsm_kernel(const float* __restrict__ S, const float* __restrict__ Cmask,
                             bf16* __restrict__ S2T) {
    int row = blockIdx.x * 8 + (threadIdx.x >> 5);       // b*LQ + q
    int lane = threadIdx.x & 31;
    int b = row / LQ, q = row % LQ;
    const float* s = S + (size_t)b * LC * LQ + q;        // stride LQ along c
    float v[16], mx = NEG_INF_F;
    #pragma unroll
    for (int i = 0; i < 16; ++i) {
        int c = lane + 32 * i;
        float cm = Cmask[b * LC + c];
        v[i] = s[(size_t)c * LQ] * cm + (1.f - cm) * NEG_INF_F;
        mx = fmaxf(mx, v[i]);
    }
    mx = wave_max(mx);
    float sum = 0.f;
    #pragma unroll
    for (int i = 0; i < 16; ++i) { v[i] = __expf(v[i] - mx); sum += v[i]; }
    sum = wave_sum(sum);
    float inv = 1.f / sum;
    bf16* o = S2T + (size_t)b * LQ * LC;                 // swizzled B of T GEMM (K=Lq, N=Lc)
    #pragma unroll
    for (int i = 0; i < 16; ++i) {
        int c = lane + 32 * i;
        o[bswz(q, c, LC)] = (bf16)(v[i] * inv);
    }
}

// ---------------- A = S1 @ Qt ; writes out[H..2H) = A, out[2H..3H) = Ct*A ------
__global__ void gemmA_kernel(const bf16* __restrict__ S1, const bf16* __restrict__ Qt,
                             const float* __restrict__ C, float* __restrict__ out) {
    const int b = blockIdx.z;
    const int m0 = blockIdx.x * 64;                      // c tile
    const int n0 = blockIdx.y * 64;                      // h tile
    const int wave = threadIdx.x >> 5;
    const int lane = threadIdx.x & 31;
    const bf16* A  = S1 + (size_t)b * LC * LQ;
    const bf16* Bm = Qt + (size_t)b * LQ * H_;
    const int mw = m0 + wave * 16;

    v8f acc[4] = {};
    #pragma unroll
    for (int k0 = 0; k0 < LQ; k0 += 32) {
        v16bf a = load_a_frag(A, LQ, mw, k0);
        #pragma unroll
        for (int t = 0; t < 4; ++t) {
            v16bf bb = load_b_frag_sw(Bm, H_, k0, n0 + t * 16);
            acc[t] = WMMA_BF16(a, bb, acc[t]);
        }
    }
    const int mh = (lane >> 4) * 8, nl = lane & 15;
    float* ob = out + (size_t)b * 4 * H_ * LC;
    #pragma unroll
    for (int t = 0; t < 4; ++t) {
        #pragma unroll
        for (int r = 0; r < 8; ++r) {
            int c = mw + r + mh;
            int h = n0 + t * 16 + nl;
            float a  = acc[t][r];
            float cv = C[((size_t)b * H_ + h) * LC + c];
            ob[(size_t)(H_ + h) * LC + c]     = a;        // A
            ob[(size_t)(2 * H_ + h) * LC + c] = cv * a;   // Ct * A
        }
    }
}

// ---------------- T = S1 @ S2^T (bf16, row-major A operand of Bm GEMM) --------
__global__ void gemmT_kernel(const bf16* __restrict__ S1, const bf16* __restrict__ S2T,
                             bf16* __restrict__ Tb) {
    const int b = blockIdx.z;
    const int m0 = blockIdx.x * 64;                      // c tile
    const int n0 = blockIdx.y * 64;                      // d tile
    const int wave = threadIdx.x >> 5;
    const int lane = threadIdx.x & 31;
    const bf16* A  = S1  + (size_t)b * LC * LQ;
    const bf16* Bm = S2T + (size_t)b * LQ * LC;
    const int mw = m0 + wave * 16;

    v8f acc[4] = {};
    #pragma unroll
    for (int k0 = 0; k0 < LQ; k0 += 32) {
        v16bf a = load_a_frag(A, LQ, mw, k0);
        #pragma unroll
        for (int t = 0; t < 4; ++t) {
            v16bf bb = load_b_frag_sw(Bm, LC, k0, n0 + t * 16);
            acc[t] = WMMA_BF16(a, bb, acc[t]);
        }
    }
    const int mh = (lane >> 4) * 8, nl = lane & 15;
    #pragma unroll
    for (int t = 0; t < 4; ++t) {
        #pragma unroll
        for (int r = 0; r < 8; ++r) {
            int m = mw + r + mh;
            int n = n0 + t * 16 + nl;
            Tb[((size_t)b * LC + m) * LC + n] = (bf16)acc[t][r];
        }
    }
}

// ---- Bm = T @ Ct with TDM-staged B panel in LDS ----
// B panel for one block: rows kt=0..15, each 2048 contiguous bf16 (4 n-tiles)
// in the swizzled Ct. One tensor_load_to_lds moves the whole 64 KB panel.
__global__ void __launch_bounds__(128)
gemmBm_kernel(const bf16* __restrict__ Tb, const bf16* __restrict__ Ctb,
              const float* __restrict__ C, float* __restrict__ out) {
    extern __shared__ bf16 smem[];                       // 16 * 2048 bf16 = 64 KB
    const int b = blockIdx.z;
    const int m0 = blockIdx.x * 64;                      // c tile
    const int n0 = blockIdx.y * 64;                      // h tile
    const int wave = threadIdx.x >> 5;
    const int lane = threadIdx.x & 31;
    const bf16* A = Tb + (size_t)b * LC * LC;
    const int mw = m0 + wave * 16;

    if (threadIdx.x < 32) {
        // D# group 0: count=1 | lds_addr | global_addr[56:0] | type=2
        const bf16* gsrc = Ctb + (size_t)b * LC * H_ + (size_t)(n0 >> 4) * 512;
        unsigned long long ga = (unsigned long long)(const void*)gsrc;
        unsigned lds = (unsigned)(unsigned long long)(const void*)smem;
        u32x4 g0;
        g0[0] = 1u;                                      // count=1, user mode
        g0[1] = lds;                                     // lds_addr
        g0[2] = (unsigned)(ga & 0xffffffffu);            // global_addr[31:0]
        g0[3] = ((unsigned)(ga >> 32) & 0x01ffffffu) | 0x80000000u; // addr[56:32] | type=2
        // D# group 1: data_size=2B; tensor row = (H/16)*512 = 24576 elements;
        // 16 rows; tile = 2048 x 16; row stride = 24576.
        i32x8 g1;
        g1[0] = 0x00010000;                              // data_size=1 (2 bytes)
        g1[1] = (int)(24576u << 16);                     // tensor_dim0 low16 -> [63:48]
        g1[2] = (int)(16u << 16);                        // dim0 hi=0, tensor_dim1=16
        g1[3] = (int)(2048u << 16);                      // dim1 hi=0, tile_dim0=2048
        g1[4] = 16;                                      // tile_dim1=16, tile_dim2=0
        g1[5] = 24576;                                   // tensor_dim0_stride low32
        g1[6] = 0;                                       // stride hi, dim1_stride low
        g1[7] = 0;
        i32x4 g2 = {0, 0, 0, 0};
        i32x4 g3 = {0, 0, 0, 0};
        asm volatile("tensor_load_to_lds %0, %1, %2, %3"
                     :: "s"(g0), "s"(g1), "s"(g2), "s"(g3)
                     : "memory");
        __builtin_amdgcn_s_wait_tensorcnt(0);
    }
    __syncthreads();

    v8f acc[4] = {};
    for (int k0 = 0; k0 < LC; k0 += 32) {
        if (k0 + 32 < LC)
            __builtin_prefetch(A + (size_t)mw * LC + k0 + 32, 0, 0);
        v16bf a = load_a_frag(A, LC, mw, k0);
        const int kt = k0 >> 5;
        #pragma unroll
        for (int t = 0; t < 4; ++t) {
            v16bf bb = load_b_frag_lds(smem, kt, t);
            acc[t] = WMMA_BF16(a, bb, acc[t]);
        }
    }
    const int mh = (lane >> 4) * 8, nl = lane & 15;
    float* ob = out + (size_t)b * 4 * H_ * LC;
    #pragma unroll
    for (int t = 0; t < 4; ++t) {
        #pragma unroll
        for (int r = 0; r < 8; ++r) {
            int c = mw + r + mh;
            int h = n0 + t * 16 + nl;
            float cv = C[((size_t)b * H_ + h) * LC + c];
            ob[(size_t)h * LC + c]            = cv;               // Ct copy
            ob[(size_t)(3 * H_ + h) * LC + c] = cv * acc[t][r];   // Ct * Bm
        }
    }
}

// ---------------- host side ----------------
extern "C" void kernel_launch(void* const* d_in, const int* in_sizes, int n_in,
                              void* d_out, int out_size, void* d_ws, size_t ws_size,
                              hipStream_t stream) {
    const float* C     = (const float*)d_in[0];
    const float* Q     = (const float*)d_in[1];
    const float* Cmask = (const float*)d_in[2];
    const float* Qmask = (const float*)d_in[3];
    const float* w4C   = (const float*)d_in[4];
    const float* w4Q   = (const float*)d_in[5];
    const float* w4mlu = (const float*)d_in[6];
    const float* bias  = (const float*)d_in[7];
    float* out = (float*)d_out;

    char* ws = (char*)d_ws;
    size_t off = 0;
    bf16*  Ctb = (bf16*)(ws + off);  off += (size_t)B_ * LC * H_ * 2;   // 25 MB (swizzled)
    bf16*  Csb = (bf16*)(ws + off);  off += (size_t)B_ * LC * H_ * 2;   // 25 MB
    bf16*  Qb  = (bf16*)(ws + off);  off += (size_t)B_ * H_ * LQ * 2;   // 6 MB (swizzled)
    bf16*  Qtb = (bf16*)(ws + off);  off += (size_t)B_ * LQ * H_ * 2;   // 6 MB (swizzled)
    float* s0  = (float*)(ws + off); off += (size_t)B_ * LC * 4;
    float* s1  = (float*)(ws + off); off += (size_t)B_ * LQ * 4;
    float* S   = (float*)(ws + off); off += (size_t)B_ * LC * LQ * 4;   // 8 MB
    bf16*  S1b = (bf16*)(ws + off);  off += (size_t)B_ * LC * LQ * 2;   // 4 MB
    bf16*  S2T = (bf16*)(ws + off);  off += (size_t)B_ * LQ * LC * 2;   // 4 MB (swizzled)
    bf16*  Tb  = (bf16*)(ws + off);  off += (size_t)B_ * LC * LC * 2;   // 16 MB

    convC_kernel<<<(B_ * H_ * LC) / 256, 256, 0, stream>>>(C, w4mlu, Ctb, Csb);
    convQ_kernel<<<(B_ * H_ * LQ) / 256, 256, 0, stream>>>(Q, Qb, Qtb);
    ks0_kernel<<<(B_ * LC) / 8, 256, 0, stream>>>(C, w4C, s0);
    ks1_kernel<<<(B_ * LQ) / 8, 256, 0, stream>>>(Q, w4Q, s1);

    gemmS_kernel<<<dim3(LC / 64, LQ / 64, B_), 128, 0, stream>>>(Csb, Qb, s0, s1, bias, S);

    rowsm_kernel<<<(B_ * LC) / 8, 256, 0, stream>>>(S, Qmask, S1b);
    colsm_kernel<<<(B_ * LQ) / 8, 256, 0, stream>>>(S, Cmask, S2T);

    gemmA_kernel<<<dim3(LC / 64, H_ / 64, B_), 128, 0, stream>>>(S1b, Qtb, C, out);
    gemmT_kernel<<<dim3(LC / 64, LC / 64, B_), 128, 0, stream>>>(S1b, S2T, Tb);
    gemmBm_kernel<<<dim3(LC / 64, H_ / 64, B_), 128, 65536, stream>>>(Tb, Ctb, C, out);
}
